// StabilityPredictorSchnet_71502615544264
// MI455X (gfx1250) — matrix operand order repaired and checked by reading, not compiled, not run on hardware
//
#include <hip/hip_runtime.h>
#include <hip/hip_bf16.h>

// Problem dims
#define BB 16
#define LL 512
#define KN 48    // neighbors
#define EE 128   // edge feature dim
#define DD 384   // node dim
#define HH 128   // head dim
// LDS pitches (in bf16 elements) chosen so every row start is 16B aligned
#define EP 136   // 136*2 = 272 = 17*16
#define TP 392   // 392*2 = 784 = 49*16

typedef __attribute__((ext_vector_type(16))) __bf16 v16bf;
typedef __attribute__((ext_vector_type(8)))  float  v8f;

union FragBF {
  v16bf v;
  uint4 q[2];
};

__device__ __forceinline__ unsigned short f2bf(float x) {
  unsigned int u = __float_as_uint(x);
  u += 0x7FFFu + ((u >> 16) & 1u);          // round-to-nearest-even
  return (unsigned short)(u >> 16);
}

__device__ __forceinline__ float gelu_erf(float x) {
  return 0.5f * x * (1.0f + erff(x * 0.7071067811865476f));
}

// A fragment: 16x32 bf16 tile from LDS (row-major, `pitch` elements per row).
// ISA layout: lanes 0-15 -> M=lane, halves 0..7 = K=k0..k0+7, halves 8..15 = K=k0+16..k0+23
//             lanes 16-31 -> M=lane-16, K offset +8 within each 16-group.
__device__ __forceinline__ FragBF load_a_lds(const unsigned short* s, int pitch,
                                             int m0, int k0, int lane) {
  const int lr = lane & 15, hi = lane >> 4;
  const unsigned short* p = s + (m0 + lr) * pitch + k0 + hi * 8;
  FragBF f;
  f.q[0] = *reinterpret_cast<const uint4*>(p);
  f.q[1] = *reinterpret_cast<const uint4*>(p + 16);
  return f;
}

// B fragment: 32x16 bf16 tile from global row-major [Kdim][ncols].
// ISA layout: N = n0 + (lane&15); halves 0..7 = K=k0+8*hi.., halves 8..15 = K=k0+16+8*hi..
__device__ __forceinline__ FragBF load_b_glb(const unsigned short* __restrict__ g,
                                             int ncols, int k0, int n0, int lane) {
  const int lr = lane & 15, hi = lane >> 4;
  const unsigned short* p = g + (size_t)(k0 + hi * 8) * ncols + n0 + lr;
  FragBF f;
  unsigned short* u = reinterpret_cast<unsigned short*>(&f);
#pragma unroll
  for (int s = 0; s < 8; ++s) {
    u[s]     = p[(size_t)s * ncols];
    u[8 + s] = p[(size_t)(16 + s) * ncols];
  }
  return f;
}

// ---------------------------------------------------------------------------
// Kernel 0: convert fw1/fw2 f32 -> bf16 once (they then live in 192MB L2).
// ---------------------------------------------------------------------------
__global__ void prep_weights(const float* __restrict__ fw1, const float* __restrict__ fw2,
                             unsigned short* __restrict__ fw1b, unsigned short* __restrict__ fw2b) {
  int i = blockIdx.x * blockDim.x + threadIdx.x;
  if (i < EE * DD) fw1b[i] = f2bf(fw1[i]);
  if (i < DD * DD) fw2b[i] = f2bf(fw2[i]);
}

// ---------------------------------------------------------------------------
// Kernel 1: fused CFConv + head. One workgroup (8 wave32) per (b,l).
//   GEMM1 (48x128 @ 128x384) -> gelu -> LDS -> GEMM2 (48x384 @ 384x384)
//   -> gelu -> multiply with gathered h_V rows, reduce over K -> MLP head -> dG
// ---------------------------------------------------------------------------
__global__ __launch_bounds__(256)
void cfconv_fused(const float* __restrict__ hV, const float* __restrict__ hE,
                  const unsigned short* __restrict__ fw1b, const float* __restrict__ fb1,
                  const unsigned short* __restrict__ fw2b, const float* __restrict__ fb2,
                  const float* __restrict__ hw1, const float* __restrict__ hb1,
                  const float* __restrict__ hw2, const float* __restrict__ hb2,
                  const float* __restrict__ hw3, const float* __restrict__ hb3,
                  const int* __restrict__ Eidx, float* __restrict__ dG) {
  __shared__ __attribute__((aligned(16))) unsigned short sA[KN * EP]; // h_E tile, bf16
  __shared__ __attribute__((aligned(16))) unsigned short sT[KN * TP]; // gelu(GEMM1), bf16
  __shared__ float sH[DD];
  __shared__ float sY1[HH];
  __shared__ float sY2[HH / 2];
  __shared__ int   sIdx[KN];

  const int bl   = blockIdx.x;          // b*512 + l
  const int b    = bl >> 9;
  const int tid  = threadIdx.x;
  const int lane = tid & 31;
  const int wv   = tid >> 5;            // 8 waves
  const int nbase = wv * 48;            // each wave owns 48 output columns
  const int lr   = lane & 15;
  const int hi   = lane >> 4;

  if (tid < KN) sIdx[tid] = Eidx[(size_t)bl * KN + tid];

  // Stage h_E[b,l] (48x128 f32) -> LDS bf16
  const float* he = hE + (size_t)bl * KN * EE;
  for (int i = tid; i < KN * EE; i += 256) {
    int r = i >> 7, c = i & (EE - 1);
    sA[r * EP + c] = f2bf(he[i]);
  }
  __syncthreads();

  const v8f vzero = {0.f, 0.f, 0.f, 0.f, 0.f, 0.f, 0.f, 0.f};
  v8f acc[3][3];
#pragma unroll
  for (int mi = 0; mi < 3; ++mi)
#pragma unroll
    for (int ni = 0; ni < 3; ++ni) acc[mi][ni] = vzero;

  // ---- GEMM1: [48x128] x [128x384] ----
  for (int k0 = 0; k0 < EE; k0 += 32) {
    FragBF a0 = load_a_lds(sA, EP,  0, k0, lane);
    FragBF a1 = load_a_lds(sA, EP, 16, k0, lane);
    FragBF a2 = load_a_lds(sA, EP, 32, k0, lane);
#pragma unroll
    for (int ni = 0; ni < 3; ++ni) {
      FragBF bm = load_b_glb(fw1b, DD, k0, nbase + ni * 16, lane);
      acc[0][ni] = __builtin_amdgcn_wmma_f32_16x16x32_bf16(false, a0.v, false, bm.v,
                                                           (short)0, acc[0][ni], false, false);
      acc[1][ni] = __builtin_amdgcn_wmma_f32_16x16x32_bf16(false, a1.v, false, bm.v,
                                                           (short)0, acc[1][ni], false, false);
      acc[2][ni] = __builtin_amdgcn_wmma_f32_16x16x32_bf16(false, a2.v, false, bm.v,
                                                           (short)0, acc[2][ni], false, false);
    }
  }

  // Epilogue 1: +bias, gelu, -> LDS bf16  (C layout: n = lane&15, m = hi*8 + r)
#pragma unroll
  for (int ni = 0; ni < 3; ++ni) {
    const int ng = nbase + ni * 16 + lr;
    const float bias = fb1[ng];
#pragma unroll
    for (int mi = 0; mi < 3; ++mi)
#pragma unroll
      for (int r = 0; r < 8; ++r) {
        int m = mi * 16 + hi * 8 + r;
        sT[m * TP + ng] = f2bf(gelu_erf(acc[mi][ni][r] + bias));
      }
  }
  __syncthreads();

#pragma unroll
  for (int mi = 0; mi < 3; ++mi)
#pragma unroll
    for (int ni = 0; ni < 3; ++ni) acc[mi][ni] = vzero;

  // ---- GEMM2: [48x384] x [384x384] ----
  for (int k0 = 0; k0 < DD; k0 += 32) {
    FragBF a0 = load_a_lds(sT, TP,  0, k0, lane);
    FragBF a1 = load_a_lds(sT, TP, 16, k0, lane);
    FragBF a2 = load_a_lds(sT, TP, 32, k0, lane);
#pragma unroll
    for (int ni = 0; ni < 3; ++ni) {
      FragBF bm = load_b_glb(fw2b, DD, k0, nbase + ni * 16, lane);
      acc[0][ni] = __builtin_amdgcn_wmma_f32_16x16x32_bf16(false, a0.v, false, bm.v,
                                                           (short)0, acc[0][ni], false, false);
      acc[1][ni] = __builtin_amdgcn_wmma_f32_16x16x32_bf16(false, a1.v, false, bm.v,
                                                           (short)0, acc[1][ni], false, false);
      acc[2][ni] = __builtin_amdgcn_wmma_f32_16x16x32_bf16(false, a2.v, false, bm.v,
                                                           (short)0, acc[2][ni], false, false);
    }
  }

  // Epilogue 2: +bias, gelu, multiply with gathered h_V rows, reduce over K=48.
  const float* hVb = hV + (size_t)b * LL * DD;
#pragma unroll
  for (int ni = 0; ni < 3; ++ni) {
    const int ng = nbase + ni * 16 + lr;
    const float bias = fb2[ng];
    float s = 0.f;
#pragma unroll
    for (int mi = 0; mi < 3; ++mi)
#pragma unroll
      for (int r = 0; r < 8; ++r) {
        int m = mi * 16 + hi * 8 + r;
        float w  = gelu_erf(acc[mi][ni][r] + bias);
        float xj = hVb[(size_t)sIdx[m] * DD + ng];  // L2-resident gather
        s += w * xj;
      }
    s += __shfl_xor(s, 16, 32);   // fold lane-pair (covers m%16 in 0..7 vs 8..15)
    if (hi == 0) sH[ng] = s;      // each wave owns a distinct 48-column slice
  }
  __syncthreads();

  // ---- MLP head (tiny; plain VALU) ----
  if (tid < HH) {
    float a = hb1[tid];
    for (int d = 0; d < DD; ++d) a += sH[d] * hw1[d * HH + tid];
    sY1[tid] = gelu_erf(a);
  }
  __syncthreads();
  if (tid < HH / 2) {
    float a = hb2[tid];
    for (int d = 0; d < HH; ++d) a += sY1[d] * hw2[d * (HH / 2) + tid];
    sY2[tid] = gelu_erf(a);
  }
  __syncthreads();
  if (tid == 0) {
    float g = hb3[0];
    for (int j = 0; j < HH / 2; ++j) g += sY2[j] * hw3[j];
    dG[bl] = g;
  }
}

// ---------------------------------------------------------------------------
// Kernel 2: masked batch reduction: out[b] = sum_l dG*mask / sqrt(max(sum mask,1))
// ---------------------------------------------------------------------------
__global__ void finalize_k(const float* __restrict__ dG, const float* __restrict__ mask,
                           float* __restrict__ out) {
  __shared__ float s1[256];
  __shared__ float s2[256];
  const int b = blockIdx.x, tid = threadIdx.x;
  float a = 0.f, m = 0.f;
  for (int l = tid; l < LL; l += 256) {
    float mk = mask[b * LL + l];
    a += dG[b * LL + l] * mk;
    m += mk;
  }
  s1[tid] = a; s2[tid] = m;
  __syncthreads();
  for (int st = 128; st > 0; st >>= 1) {
    if (tid < st) { s1[tid] += s1[tid + st]; s2[tid] += s2[tid + st]; }
    __syncthreads();
  }
  if (tid == 0) out[b] = s1[0] / sqrtf(fmaxf(s2[0], 1.0f));
}

extern "C" void kernel_launch(void* const* d_in, const int* in_sizes, int n_in,
                              void* d_out, int out_size, void* d_ws, size_t ws_size,
                              hipStream_t stream) {
  (void)in_sizes; (void)n_in; (void)out_size; (void)ws_size;
  const float* hV   = (const float*)d_in[0];
  const float* hE   = (const float*)d_in[1];
  const float* mask = (const float*)d_in[2];
  const float* fw1  = (const float*)d_in[3];
  const float* fb1  = (const float*)d_in[4];
  const float* fw2  = (const float*)d_in[5];
  const float* fb2  = (const float*)d_in[6];
  const float* hw1  = (const float*)d_in[7];
  const float* hb1  = (const float*)d_in[8];
  const float* hw2  = (const float*)d_in[9];
  const float* hb2  = (const float*)d_in[10];
  const float* hw3  = (const float*)d_in[11];
  const float* hb3  = (const float*)d_in[12];
  const int*   Eidx = (const int*)d_in[13];

  // Workspace layout: fw1 bf16 (98304 B) | fw2 bf16 (294912 B) | dG f32 (32768 B)
  unsigned short* fw1b = (unsigned short*)d_ws;
  unsigned short* fw2b = fw1b + EE * DD;
  float* dG = (float*)(fw2b + DD * DD);

  prep_weights<<<(DD * DD + 255) / 256, 256, 0, stream>>>(fw1, fw2, fw1b, fw2b);
  cfconv_fused<<<BB * LL, 256, 0, stream>>>(hV, hE, fw1b, fb1, fw2b, fb2,
                                            hw1, hb1, hw2, hb2, hw3, hb3, Eidx, dG);
  finalize_k<<<BB, 256, 0, stream>>>(dG, mask, (float*)d_out);
}
